// FeatureTransformer_9002251452818
// MI455X (gfx1250) — matrix operand order repaired and checked by reading, not compiled, not run on hardware
//
#include <hip/hip_runtime.h>
#include <hip/hip_fp16.h>
#include <math.h>

typedef __attribute__((ext_vector_type(16))) _Float16 v16h;
typedef __attribute__((ext_vector_type(8)))  _Float16 v8h;
typedef __attribute__((ext_vector_type(4)))  _Float16 v4h;
typedef __attribute__((ext_vector_type(8)))  float    v8f;

#define NROW   24576      // 4 * 64 * 96 tokens
#define CDIM   128
#define LDA    136        // f16 row stride for 128-wide LDS tiles (272B = 16*17)
#define LDV    40         // f16 row stride for 32-wide LDS tiles (80B = 16*5)
#define SCALE  0.0883883476483184f   // 1/sqrt(128)

// ---------------------------------------------------------------- fragments
__device__ __forceinline__ v16h frag_a(const _Float16* p) {
  // A 16x32 f16: lanes 0-15 row=l16 K{0..15}; lanes 16-31 K{16..31} (contiguous)
  v16h r;
  *((v8h*)&r)       = *((const v8h*)(p));
  *(((v8h*)&r) + 1) = *((const v8h*)(p + 8));
  return r;
}
__device__ __forceinline__ v16h frag_b(const _Float16* p) {
  // B 32x16 f16: lanes 0-15 col=l16 K{0..15}; lanes 16-31 K{16..31} (contiguous)
  v16h r;
  *((v8h*)&r)       = *((const v8h*)(p));
  *(((v8h*)&r) + 1) = *((const v8h*)(p + 8));
  return r;
}
__device__ __forceinline__ v4h pk4(float a, float b, float c, float d) {
  v4h h;
  h[0] = (_Float16)a; h[1] = (_Float16)b;
  h[2] = (_Float16)c; h[3] = (_Float16)d;
  return h;
}
__device__ __forceinline__ float hmax16(float v) {
  v = fmaxf(v, __shfl_xor(v, 1, 32));
  v = fmaxf(v, __shfl_xor(v, 2, 32));
  v = fmaxf(v, __shfl_xor(v, 4, 32));
  v = fmaxf(v, __shfl_xor(v, 8, 32));
  return v;
}
__device__ __forceinline__ float hsum16(float v) {
  v += __shfl_xor(v, 1, 32); v += __shfl_xor(v, 2, 32);
  v += __shfl_xor(v, 4, 32); v += __shfl_xor(v, 8, 32);
  return v;
}
__device__ __forceinline__ float hsum32(float v) {
  v = hsum16(v); v += __shfl_xor(v, 16, 32);
  return v;
}

// ------------------------------------------------------------- GEMM (WMMA)
// C[M,N] = A[M,K] @ W[K,N]; A row read as (row+row_swap)%M; act=1 -> exact GELU
__global__ __launch_bounds__(256) void k_gemm(
    const float* __restrict__ A, const float* __restrict__ W,
    float* __restrict__ C, int M, int N, int K, int row_swap, int act)
{
  __shared__ _Float16 As[128 * LDA];   // A tile, row-major [128m][128k]
  __shared__ _Float16 Ws[64 * LDA];    // W tile transposed [64n][128k]
  const int tid = threadIdx.x;
  const int bm = blockIdx.x, bn = blockIdx.y;
  const int wave = tid >> 5, lane = tid & 31, hf = lane >> 4, l16 = lane & 15;
  v8f acc[4] = {};

  for (int kk = 0; kk < K; kk += 128) {
    #pragma unroll
    for (int i = 0; i < 16; i++) {                 // stage A (4096 float4 -> b64 stores)
      int f4 = i * 256 + tid;
      int r = f4 >> 5, c4 = f4 & 31;
      int gr = bm * 128 + r + row_swap; if (gr >= M) gr -= M;
      const float4 v = *(const float4*)(A + (size_t)gr * K + kk + c4 * 4);
      *(v4h*)(&As[r * LDA + c4 * 4]) = pk4(v.x, v.y, v.z, v.w);
    }
    #pragma unroll
    for (int i = 0; i < 8; i++) {                  // stage W transposed, gather-4 along K
      int fl = i * 256 + tid;                      // 2048 items: 64n x 32 k-quads
      int n = fl & 63, k4 = fl >> 6;
      const float* wp = W + (size_t)(kk + k4 * 4) * N + bn * 64 + n;
      *(v4h*)(&Ws[n * LDA + k4 * 4]) =
          pk4(wp[0], wp[N], wp[2 * (size_t)N], wp[3 * (size_t)N]);
    }
    __syncthreads();
    #pragma unroll
    for (int kc = 0; kc < 4; kc++) {
      v16h a = frag_a(&As[(wave * 16 + l16) * LDA + kc * 32 + hf * 16]);
      #pragma unroll
      for (int nt = 0; nt < 4; nt++) {
        v16h b = frag_b(&Ws[(nt * 16 + l16) * LDA + kc * 32 + hf * 16]);
        acc[nt] = __builtin_amdgcn_wmma_f32_16x16x32_f16(
            false, a, false, b, (short)0, acc[nt], false, false);
      }
    }
    __syncthreads();
  }
  #pragma unroll
  for (int nt = 0; nt < 4; nt++) {
    int col = bn * 64 + nt * 16 + l16;
    #pragma unroll
    for (int r = 0; r < 8; r++) {
      int row = bm * 128 + wave * 16 + hf * 8 + r;
      float v = acc[nt][r];
      if (act) v = 0.5f * v * (1.0f + erff(v * 0.70710678118654752f));
      C[(size_t)row * N + col] = v;
    }
  }
}

// ------------------------------------------------- 2D shifted-window attn
__device__ __forceinline__ int tok_row(int bb, int ky, int kx, int t, int shift) {
  int wy = t / 48, wx = t - wy * 48;
  int y = ky * 32 + wy, x = kx * 48 + wx;
  if (shift) { y = (y + 16) & 63; x += 24; if (x >= 96) x -= 96; }
  return bb * 6144 + y * 96 + x;
}
__device__ __forceinline__ int region2d(int ky, int kx, int t) {
  int wy = t / 48, wx = t - wy * 48;
  int y = ky * 32 + wy, x = kx * 48 + wx;
  int ry = (y < 32) ? 0 : ((y < 48) ? 1 : 2);
  int rx = (x < 48) ? 0 : ((x < 72) ? 1 : 2);
  return ry * 3 + rx;
}

// grid: 16 windows * 12 query-blocks; block 256 = 8 waves * 16 query rows
__global__ __launch_bounds__(256) void k_attn2d(
    const float* __restrict__ Q, const float* __restrict__ Kg,
    const float* __restrict__ Vg, float* __restrict__ O, int shift)
{
  __shared__ _Float16 Qs[128 * LDA];     // 34816 B
  __shared__ _Float16 Ks[32 * LDA];      //  8704 B
  __shared__ _Float16 Vt[128 * LDV];     // 10240 B  (V transposed: [c][key])
  __shared__ _Float16 Ps[8 * 16 * LDV];  // 10240 B  (per-wave P staging)
  const int tid = threadIdx.x, wave = tid >> 5, lane = tid & 31;
  const int hf = lane >> 4, l16 = lane & 15;
  const int wid = blockIdx.x / 12, qtb = blockIdx.x - wid * 12;
  const int bb = wid >> 2, ky = (wid >> 1) & 1, kx = wid & 1;

  #pragma unroll
  for (int i = 0; i < 16; i++) {                   // stage 128 query rows
    int f4 = i * 256 + tid;
    int r = f4 >> 5, c4 = f4 & 31;
    int grow = tok_row(bb, ky, kx, qtb * 128 + r, shift);
    const float4 v = *(const float4*)(Q + (size_t)grow * 128 + c4 * 4);
    *(v4h*)(&Qs[r * LDA + c4 * 4]) = pk4(v.x, v.y, v.z, v.w);
  }
  __syncthreads();
  v16h qf[4];
  #pragma unroll
  for (int kc = 0; kc < 4; kc++)
    qf[kc] = frag_a(&Qs[(wave * 16 + l16) * LDA + kc * 32 + hf * 16]);

  v8f o[8] = {};
  float mrow[8], lrow[8];
  #pragma unroll
  for (int r = 0; r < 8; r++) { mrow[r] = -1e30f; lrow[r] = 0.f; }
  _Float16* Pw = &Ps[wave * 16 * LDV];

  for (int kt = 0; kt < 48; kt++) {                // 1536 keys / 32
    #pragma unroll
    for (int i = 0; i < 4; i++) {                  // stage 32 key rows (row-major)
      int f4 = i * 256 + tid;
      int kkr = f4 >> 5, c4 = f4 & 31;
      int grow = tok_row(bb, ky, kx, kt * 32 + kkr, shift);
      const float4 kv = *(const float4*)(Kg + (size_t)grow * 128 + c4 * 4);
      *(v4h*)(&Ks[kkr * LDA + c4 * 4]) = pk4(kv.x, kv.y, kv.z, kv.w);
    }
    #pragma unroll
    for (int i = 0; i < 4; i++) {                  // stage V transposed, gather-4 keys
      int fl = i * 256 + tid;                      // 1024 items: 128c x 8 key-quads
      int c = fl & 127, k4 = fl >> 7;
      v4h h;
      #pragma unroll
      for (int j = 0; j < 4; j++) {
        int grow = tok_row(bb, ky, kx, kt * 32 + k4 * 4 + j, shift);
        h[j] = (_Float16)Vg[(size_t)grow * 128 + c];
      }
      *(v4h*)(&Vt[c * LDV + k4 * 4]) = h;
    }
    __syncthreads();

    v8f s0 = {}, s1 = {};
    #pragma unroll
    for (int kc = 0; kc < 4; kc++) {               // S = Q K^T (K rows are B cols)
      v16h b0 = frag_b(&Ks[l16 * LDA + kc * 32 + hf * 16]);
      v16h b1 = frag_b(&Ks[(16 + l16) * LDA + kc * 32 + hf * 16]);
      s0 = __builtin_amdgcn_wmma_f32_16x16x32_f16(false, qf[kc], false, b0, (short)0, s0, false, false);
      s1 = __builtin_amdgcn_wmma_f32_16x16x32_f16(false, qf[kc], false, b1, (short)0, s1, false, false);
    }
    int idk0 = 0, idk1 = 0;
    if (shift) {
      idk0 = region2d(ky, kx, kt * 32 + l16);
      idk1 = region2d(ky, kx, kt * 32 + 16 + l16);
    }
    #pragma unroll
    for (int r = 0; r < 8; r++) {                  // online softmax per row
      float a0 = s0[r] * SCALE, a1 = s1[r] * SCALE;
      if (shift) {
        int idq = region2d(ky, kx, qtb * 128 + wave * 16 + hf * 8 + r);
        if (idq != idk0) a0 -= 100.f;
        if (idq != idk1) a1 -= 100.f;
      }
      float mt = hmax16(fmaxf(a0, a1));
      float mn = fmaxf(mrow[r], mt);
      float corr = __expf(mrow[r] - mn);
      float p0 = __expf(a0 - mn), p1 = __expf(a1 - mn);
      lrow[r] = lrow[r] * corr + hsum16(p0 + p1);
      mrow[r] = mn;
      #pragma unroll
      for (int nt = 0; nt < 8; nt++) o[nt][r] *= corr;
      Pw[(hf * 8 + r) * LDV + l16]      = (_Float16)p0;
      Pw[(hf * 8 + r) * LDV + 16 + l16] = (_Float16)p1;
    }
    __builtin_amdgcn_wave_barrier();
    asm volatile("s_wait_dscnt 0" ::: "memory");   // wave-internal P store->load
    __builtin_amdgcn_wave_barrier();
    v16h pf = frag_a(&Pw[l16 * LDV + hf * 16]);
    #pragma unroll
    for (int nt = 0; nt < 8; nt++) {               // O += P V
      v16h vf = frag_b(&Vt[(nt * 16 + l16) * LDV + hf * 16]);
      o[nt] = __builtin_amdgcn_wmma_f32_16x16x32_f16(false, pf, false, vf, (short)0, o[nt], false, false);
    }
    __syncthreads();
  }
  #pragma unroll
  for (int r = 0; r < 8; r++) {
    float inv = 1.f / lrow[r];
    int grow = tok_row(bb, ky, kx, qtb * 128 + wave * 16 + hf * 8 + r, shift);
    #pragma unroll
    for (int nt = 0; nt < 8; nt++)
      O[(size_t)grow * 128 + nt * 16 + l16] = o[nt][r] * inv;
  }
}

// ------------------------------------------------- 1D cross attention (tiny)
__global__ __launch_bounds__(64) void k_attn1d(
    const float* __restrict__ Q, const float* __restrict__ Kb,
    const float* __restrict__ Vb, float* __restrict__ O, int shift)
{
  __shared__ float Qs[48 * 132];
  __shared__ float Ks[48 * 128];
  __shared__ float Ss[48 * 48];
  __shared__ int   Vr[48];
  const int tid = threadIdx.x;
  const int wid = blockIdx.x;
  const int bb = wid / 128, rem = wid - bb * 128;
  const int y = rem >> 1, kxw = rem & 1;

  if (tid < 48) {
    int x = kxw * 48 + tid;
    if (shift) { x += 24; if (x >= 96) x -= 96; }
    Vr[tid] = bb * 6144 + y * 96 + x;
  }
  for (int e = tid; e < 48 * 128; e += 64) {
    int j = e >> 7, c = e & 127;
    int x = kxw * 48 + j;
    if (shift) { x += 24; if (x >= 96) x -= 96; }
    size_t row = (size_t)(bb * 6144 + y * 96 + x) * 128 + c;
    Qs[j * 132 + c] = Q[row];
    Ks[j * 128 + c] = Kb[row];
  }
  __syncthreads();
  if (tid < 48) {
    float mx = -1e30f;
    for (int j = 0; j < 48; j++) {
      float d = 0.f;
      #pragma unroll 8
      for (int c = 0; c < 128; c++) d += Qs[tid * 132 + c] * Ks[j * 128 + c];
      d *= SCALE;
      if (shift) {
        int xi = kxw * 48 + tid, xj = kxw * 48 + j;
        int ri = (xi < 48) ? 0 : ((xi < 72) ? 1 : 2);
        int rj = (xj < 48) ? 0 : ((xj < 72) ? 1 : 2);
        if (ri != rj) d -= 100.f;
      }
      Ss[tid * 48 + j] = d; mx = fmaxf(mx, d);
    }
    float sum = 0.f;
    for (int j = 0; j < 48; j++) {
      float p = __expf(Ss[tid * 48 + j] - mx);
      Ss[tid * 48 + j] = p; sum += p;
    }
    float inv = 1.f / sum;
    size_t orow = (size_t)Vr[tid] * 128;
    for (int c = 0; c < 128; c++) {
      float a = 0.f;
      for (int j = 0; j < 48; j++) a += Ss[tid * 48 + j] * Vb[(size_t)Vr[j] * 128 + c];
      O[orow + c] = a * inv;
    }
  }
}

// ------------------------------------------------- LayerNorm (+residual)
__global__ __launch_bounds__(256) void k_ln(
    float* __restrict__ dst, const float* __restrict__ src,
    const float* __restrict__ res, const float* __restrict__ g,
    const float* __restrict__ b, int addres)
{
  const int tid = threadIdx.x, wave = tid >> 5, lane = tid & 31;
  const size_t row = (size_t)blockIdx.x * 8 + wave;
  const float4 x = *(const float4*)(src + row * 128 + lane * 4);
  float mean = hsum32(x.x + x.y + x.z + x.w) * (1.f / 128.f);
  float d0 = x.x - mean, d1 = x.y - mean, d2 = x.z - mean, d3 = x.w - mean;
  float var = hsum32(d0 * d0 + d1 * d1 + d2 * d2 + d3 * d3) * (1.f / 128.f);
  float rstd = rsqrtf(var + 1e-5f);
  const float4 gv = *(const float4*)(g + lane * 4);
  const float4 bv = *(const float4*)(b + lane * 4);
  float4 yv;
  yv.x = d0 * rstd * gv.x + bv.x; yv.y = d1 * rstd * gv.y + bv.y;
  yv.z = d2 * rstd * gv.z + bv.z; yv.w = d3 * rstd * gv.w + bv.w;
  if (addres) {
    const float4 rv = *(const float4*)(res + row * 128 + lane * 4);
    yv.x += rv.x; yv.y += rv.y; yv.z += rv.z; yv.w += rv.w;
  }
  *(float4*)(dst + row * 128 + lane * 4) = yv;
}

// ------------------------------------------------- misc data movement
__global__ __launch_bounds__(256) void k_concat(
    float* __restrict__ dst, const float* __restrict__ a, const float* __restrict__ b)
{
  int i = blockIdx.x * 256 + threadIdx.x;   // float4 index over [24576][256]
  int row = i >> 6, c4 = i & 63;
  float4 v = (c4 < 32) ? *(const float4*)(a + (size_t)row * 128 + c4 * 4)
                       : *(const float4*)(b + (size_t)row * 128 + (c4 - 32) * 4);
  *(float4*)(dst + (size_t)row * 256 + c4 * 4) = v;
}

__global__ __launch_bounds__(256) void k_pack(
    float* __restrict__ c0, const float* __restrict__ f0, const float* __restrict__ f1)
{
  int i = blockIdx.x * 256 + threadIdx.x;   // 0 .. 3145727
  int bb = i / (128 * 6144);
  int r = i - bb * (128 * 6144);
  int ch = r / 6144, pos = r - ch * 6144;
  float v = (bb < 2) ? f0[(size_t)(bb * 128 + ch) * 6144 + pos]
                     : f1[(size_t)((bb - 2) * 128 + ch) * 6144 + pos];
  c0[(size_t)(bb * 6144 + pos) * 128 + ch] = v;
}

__global__ __launch_bounds__(256) void k_unpack(
    float* __restrict__ out, const float* __restrict__ c0)
{
  int i = blockIdx.x * 256 + threadIdx.x;   // 0 .. 3145727
  int sec = i / 1572864;
  int r = i - sec * 1572864;
  int bi = r / 786432;
  int r2 = r - bi * 786432;
  int ch = r2 / 6144, pos = r2 - ch * 6144;
  int batch = sec * 2 + bi;
  out[i] = c0[(size_t)(batch * 6144 + pos) * 128 + ch];
}

// ------------------------------------------------------------------ driver
static inline void gemm(hipStream_t s, const float* A, const float* W, float* C,
                        int M, int N, int K, int swap, int act) {
  dim3 g(M / 128, N / 64);
  k_gemm<<<g, 256, 0, s>>>(A, W, C, M, N, K, swap, act);
}

extern "C" void kernel_launch(void* const* d_in, const int* in_sizes, int n_in,
                              void* d_out, int out_size, void* d_ws, size_t ws_size,
                              hipStream_t stream)
{
  (void)in_sizes; (void)n_in; (void)out_size; (void)ws_size;
  const float* f0  = (const float*)d_in[0];
  const float* f1  = (const float*)d_in[1];
  const float* sq  = (const float*)d_in[2];
  const float* sk  = (const float*)d_in[3];
  const float* sv  = (const float*)d_in[4];
  const float* sm  = (const float*)d_in[5];
  const float* sg1 = (const float*)d_in[6];
  const float* sb1 = (const float*)d_in[7];
  const float* cq  = (const float*)d_in[8];
  const float* ck  = (const float*)d_in[9];
  const float* cv  = (const float*)d_in[10];
  const float* cm  = (const float*)d_in[11];
  const float* cg1 = (const float*)d_in[12];
  const float* cb1 = (const float*)d_in[13];
  const float* cw1 = (const float*)d_in[14];
  const float* cw2 = (const float*)d_in[15];
  const float* cg2 = (const float*)d_in[16];
  const float* cb2 = (const float*)d_in[17];

  float* ws = (float*)d_ws;
  const size_t SZ = (size_t)NROW * CDIM;           // 3,145,728 floats
  float* c0   = ws;
  float* qb   = c0   + SZ;
  float* kb   = qb   + SZ;
  float* vb   = kb   + SZ;
  float* msg  = vb   + SZ;
  float* tmp  = msg  + SZ;
  float* msg2 = tmp  + SZ;
  float* kc   = msg2 + SZ;
  float* vc   = kc   + SZ;
  float* cc   = vc   + SZ;                         // [24576][256]
  float* hdn  = cc + (size_t)NROW * 256;           // [6144][1024] chunk

  k_pack<<<12288, 256, 0, stream>>>(c0, f0, f1);

  for (int i = 0; i < 6; i++) {
    const int shift = (i & 1);
    const float* sq_i = sq + (size_t)i * 16384;
    const float* sk_i = sk + (size_t)i * 16384;
    const float* sv_i = sv + (size_t)i * 16384;
    const float* sm_i = sm + (size_t)i * 16384;
    const float* cq_i = cq + (size_t)i * 16384;
    const float* ck_i = ck + (size_t)i * 16384;
    const float* cv_i = cv + (size_t)i * 16384;
    const float* cm_i = cm + (size_t)i * 16384;
    const float* cw1_i = cw1 + (size_t)i * 256 * 1024;
    const float* cw2_i = cw2 + (size_t)i * 1024 * 128;

    // Cross k/v computed from c1 = swap(c0) BEFORE self updates c0
    // (reference uses c1 from end of previous iteration).
    gemm(stream, c0, ck_i, kc, NROW, 128, 128, 12288, 0);
    gemm(stream, c0, cv_i, vc, NROW, 128, 128, 12288, 0);

    // ---- self attention layer
    gemm(stream, c0, sq_i, qb, NROW, 128, 128, 0, 0);
    gemm(stream, c0, sk_i, kb, NROW, 128, 128, 0, 0);
    gemm(stream, c0, sv_i, vb, NROW, 128, 128, 0, 0);
    k_attn2d<<<192, 256, 0, stream>>>(qb, kb, vb, msg, shift);
    gemm(stream, msg, sm_i, tmp, NROW, 128, 128, 0, 0);
    k_ln<<<3072, 256, 0, stream>>>(c0, tmp, c0, sg1 + i * 128, sb1 + i * 128, 1);

    // ---- cross attention layer + FFN
    gemm(stream, c0, cq_i, qb, NROW, 128, 128, 0, 0);
    k_attn1d<<<512, 64, 0, stream>>>(qb, kc, vc, msg, shift);
    gemm(stream, msg, cm_i, tmp, NROW, 128, 128, 0, 0);
    k_ln<<<3072, 256, 0, stream>>>(msg2, tmp, nullptr, cg1 + i * 128, cb1 + i * 128, 0);
    k_concat<<<6144, 256, 0, stream>>>(cc, c0, msg2);
    for (int chk = 0; chk < 4; chk++) {            // FFN in 6144-row chunks
      gemm(stream, cc + (size_t)chk * 6144 * 256, cw1_i, hdn, 6144, 1024, 256, 0, 1);
      gemm(stream, hdn, cw2_i, tmp + (size_t)chk * 6144 * 128, 6144, 128, 1024, 0, 0);
    }
    k_ln<<<3072, 256, 0, stream>>>(c0, tmp, c0, cg2 + i * 128, cb2 + i * 128, 1);
  }

  k_unpack<<<12288, 256, 0, stream>>>((float*)d_out, c0);
}
